// SS2D_RegionLite_2_Tiny_83769042141522
// MI455X (gfx1250) — matrix operand order repaired and checked by reading, not compiled, not run on hardware
//
#include <hip/hip_runtime.h>
#include <hip/hip_bf16.h>
#include <math.h>

#define C_    192
#define S_    16
#define G_    2
#define MID_  48
#define H_    48
#define W_    48
#define B_    2
#define L_    (H_ * W_)          // 2304
#define CHW_  (C_ * H_ * W_)     // 442368
#define OC_   (2 * C_ * S_)      // 6144
#define KPAD_ 64                 // K=48 padded to 64 for 2x 16x16x32 WMMA

typedef __attribute__((ext_vector_type(16))) __bf16 v16bf;
typedef __attribute__((ext_vector_type(8)))  float  v8f;
typedef __attribute__((ext_vector_type(4)))  unsigned u32x4;
typedef __attribute__((ext_vector_type(8)))  int      i32x8;
typedef __attribute__((ext_vector_type(4)))  int      i32x4;

union BF16x16 { v16bf v; uint4 q[2]; unsigned short u[16]; };

static __device__ __forceinline__ unsigned short f2bf(float f) {
    union { float f; unsigned u; } v; v.f = f;
    unsigned r = v.u + 0x7FFFu + ((v.u >> 16) & 1u);   // round-to-nearest-even
    return (unsigned short)(r >> 16);
}
static __device__ __forceinline__ float bf2f(unsigned short h) {
    union { unsigned u; float f; } v; v.u = ((unsigned)h) << 16; return v.f;
}

// ---------------- 1. per-batch mean / rstd over (C,H,W) ----------------
__global__ void k_gn_stats(const float* __restrict__ x, float* __restrict__ st) {
    int b = blockIdx.x;
    const float* p = x + (size_t)b * CHW_;
    float s = 0.f, s2 = 0.f;
    for (int i = threadIdx.x; i < CHW_; i += blockDim.x) { float v = p[i]; s += v; s2 += v * v; }
    __shared__ float sh0[256], sh1[256];
    sh0[threadIdx.x] = s; sh1[threadIdx.x] = s2;
    __syncthreads();
    for (int off = 128; off > 0; off >>= 1) {
        if (threadIdx.x < off) { sh0[threadIdx.x] += sh0[threadIdx.x + off]; sh1[threadIdx.x] += sh1[threadIdx.x + off]; }
        __syncthreads();
    }
    if (threadIdx.x == 0) {
        float mu = sh0[0] / (float)CHW_;
        float var = sh1[0] / (float)CHW_ - mu * mu;
        st[b * 2 + 0] = mu;
        st[b * 2 + 1] = rsqrtf(var + 1e-6f);
    }
}

// ------------- 2. fused GN-affine + depthwise 3x3 + SiLU → z(B,C,L) -------------
__global__ void k_gn_conv_silu(const float* __restrict__ x, const float* __restrict__ gw,
                               const float* __restrict__ gb, const float* __restrict__ dww,
                               const float* __restrict__ dwb, const float* __restrict__ st,
                               float* __restrict__ z) {
    int idx = blockIdx.x * blockDim.x + threadIdx.x;
    if (idx >= B_ * CHW_) return;
    int w = idx % W_, h = (idx / W_) % H_, c = (idx / L_) % C_, b = idx / (C_ * L_);
    float mu = st[b * 2 + 0], rs = st[b * 2 + 1];
    float sc = rs * gw[c], sh = gb[c] - mu * sc;
    const float* xp = x + ((size_t)b * C_ + c) * L_;
    const float* wp = dww + c * 9;
    float acc = 0.f;
#pragma unroll
    for (int ky = 0; ky < 3; ++ky) {
        int hy = h + ky - 1;
        if (hy < 0 || hy >= H_) continue;
#pragma unroll
        for (int kx = 0; kx < 3; ++kx) {
            int wx = w + kx - 1;
            if (wx < 0 || wx >= W_) continue;
            acc += (xp[hy * W_ + wx] * sc + sh) * wp[ky * 3 + kx];
        }
    }
    float a = acc + dwb[c];
    z[idx] = a / (1.f + __expf(-a));
}

// ------------- 3. stage-1 low-rank (in & dt share z reads): t1[b][m][l] -------------
__global__ void k_proj1(const float* __restrict__ z, const float* __restrict__ w_in,
                        const float* __restrict__ w_dt, float* __restrict__ t1i,
                        float* __restrict__ t1d) {
    int idx = blockIdx.x * blockDim.x + threadIdx.x;
    if (idx >= B_ * MID_ * L_) return;
    int l = idx % L_, m = (idx / L_) % MID_, b = idx / (MID_ * L_);
    const float* zb = z + (size_t)b * C_ * L_ + l;
    float a1 = 0.f, a2 = 0.f;
    for (int c = 0; c < C_; ++c) {
        float zv = zb[(size_t)c * L_];
        a1 += zv * w_in[m * C_ + c];
        a2 += zv * w_dt[m * C_ + c];
    }
    t1i[idx] = a1; t1d[idx] = a2;
}

// ------------- 4. grouped bc_mid[b][gm][l] -------------
__global__ void k_bcmid(const float* __restrict__ z, const float* __restrict__ bw,
                        float* __restrict__ bcm) {
    int idx = blockIdx.x * blockDim.x + threadIdx.x;
    if (idx >= B_ * MID_ * L_) return;
    int l = idx % L_, gm = (idx / L_) % MID_, b = idx / (MID_ * L_);
    int g = gm / (MID_ / G_);
    const float* zb = z + ((size_t)b * C_ + g * (C_ / G_)) * L_ + l;
    float acc = 0.f;
    for (int c = 0; c < C_ / G_; ++c)
        acc += zb[(size_t)c * L_] * bw[gm * (C_ / G_) + c];
    bcm[idx] = acc;
}

// ------------- 5. stage-2 low-rank → u(B,C,L), delta_pre(B,C,L) -------------
__global__ void k_proj2(const float* __restrict__ t1i, const float* __restrict__ t1d,
                        const float* __restrict__ w2i, const float* __restrict__ b2i,
                        const float* __restrict__ w2d, const float* __restrict__ b2d,
                        float* __restrict__ u, float* __restrict__ d) {
    int idx = blockIdx.x * blockDim.x + threadIdx.x;
    if (idx >= B_ * C_ * L_) return;
    int l = idx % L_, co = (idx / L_) % C_, b = idx / (C_ * L_);
    const float* ti = t1i + (size_t)b * MID_ * L_ + l;
    const float* td = t1d + (size_t)b * MID_ * L_ + l;
    float au = b2i[co], ad = b2d[co];
    for (int m = 0; m < MID_; ++m) {
        au += ti[(size_t)m * L_] * w2i[co * MID_ + m];
        ad += td[(size_t)m * L_] * w2d[co * MID_ + m];
    }
    u[idx] = au; d[idx] = ad;
}

// ------------- 6/7. bf16 staging (K padded 48 → 64) -------------
__global__ void k_cvt_w(const float* __restrict__ w, unsigned short* __restrict__ Wbf) {
    int i = blockIdx.x * blockDim.x + threadIdx.x;
    if (i >= OC_ * KPAD_) return;
    int o = i >> 6, k = i & 63;
    Wbf[i] = (k < MID_) ? f2bf(w[o * MID_ + k]) : (unsigned short)0;
}
__global__ void k_cvt_x(const float* __restrict__ bcm, unsigned short* __restrict__ Xbf) {
    int i = blockIdx.x * blockDim.x + threadIdx.x;
    if (i >= B_ * L_ * KPAD_) return;
    int k = i & 63, l = (i >> 6) % L_, b = i / (KPAD_ * L_);
    Xbf[i] = (k < MID_) ? f2bf(bcm[((size_t)b * MID_ + k) * L_ + l]) : (unsigned short)0;
}

// ------------- 8. WMMA GEMM: bc_all(bf16) = bch_w @ bc_mid + bias -------------
// Block = one o-tile (16x64 bf16 A panel, 2KB) staged LDS via the Tensor Data
// Mover; 8 waves x 2 l-tiles consume it. K=64 as two v_wmma_f32_16x16x32_bf16.
__global__ void __launch_bounds__(256) k_bc_gemm_wmma(
    const unsigned short* __restrict__ Wbf,  // [OC_][64] bf16
    const unsigned short* __restrict__ Xbf,  // [B*L][64] bf16 (l-major rows)
    const float* __restrict__ bias,          // [OC_]
    unsigned short* __restrict__ bc_all) {   // [B][OC_][L] bf16
    const int OT = OC_ / 16, LT = L_ / 16;   // 384, 144
    const int LTC = LT / 16;                 // 9 blocks along L per o-tile
    int blk = blockIdx.x;
    int ltc = blk % LTC;
    int ot  = (blk / LTC) % OT;
    int b   = blk / (LTC * OT);
    int wv   = threadIdx.x >> 5;
    int lane = threadIdx.x & 31;
    int hi = lane >> 4;        // half-wave select per ISA fragment layout
    int mn = lane & 15;        // A row / B,D column

    __shared__ __attribute__((aligned(16))) unsigned char lds_a[2048]; // 16 rows x 128B

    if (threadIdx.x < 32) {
        // --- TDM descriptor (ISA 8.3/8.4): 2D tile, data_size=4B,
        //     tile 32 dwords x 16 rows, row stride 32 dwords.
        unsigned long long ga = (unsigned long long)(const void*)(Wbf + (size_t)(ot * 16) * KPAD_);
        unsigned ldsoff = (unsigned)(size_t)(void*)lds_a;   // generic addr low 32 = LDS offset
        u32x4 g0 = { 1u,                                    // count=1, user mode
                     ldsoff,                                // lds_addr
                     (unsigned)ga,                          // global_addr[31:0]
                     ((unsigned)(ga >> 32) & 0x01FFFFFFu) | 0x80000000u }; // addr[56:32] | type=2
        i32x8 g1 = { (int)(2u << 16),                       // data_size=2 (4 bytes)
                     (int)(32u << 16),                      // tensor_dim0 = 32 dwords
                     (int)(6144u << 16),                    // tensor_dim1 = 6144 rows
                     (int)(32u << 16),                      // tile_dim0 = 32 dwords
                     16,                                    // tile_dim1 = 16 rows
                     32,                                    // tensor_dim0_stride = 32 dwords
                     0, 0 };
        i32x4 z4 = { 0, 0, 0, 0 };
#if defined(__clang_major__) && (__clang_major__ >= 23)
        i32x8 z8 = { 0, 0, 0, 0, 0, 0, 0, 0 };
        __builtin_amdgcn_tensor_load_to_lds(g0, g1, z4, z4, z8, 0);
#else
        __builtin_amdgcn_tensor_load_to_lds(g0, g1, z4, z4, 0);
#endif
        __builtin_amdgcn_s_wait_tensorcnt(0);
    }
    __syncthreads();

    // A fragment from LDS: row mn; K chunks [8*hi,8*hi+8) and [16+8*hi,24+8*hi) (+k0)
    const uint4* rowp = reinterpret_cast<const uint4*>(lds_a + (size_t)mn * 128);
    BF16x16 a0, a1;
    a0.q[0] = rowp[hi];     a0.q[1] = rowp[2 + hi];      // K 0..31
    a1.q[0] = rowp[4 + hi]; a1.q[1] = rowp[6 + hi];      // K 32..63

    int lt0 = ltc * 16 + wv * 2;
    __builtin_prefetch(reinterpret_cast<const void*>(
        Xbf + ((size_t)b * L_ + (lt0 + 1) * 16 + mn) * KPAD_), 0, 1);

#pragma unroll
    for (int t = 0; t < 2; ++t) {
        int lt = lt0 + t;
        // B fragment: column lt*16+mn; K chunks [8*hi,8*hi+8) and [16+8*hi,24+8*hi) (+k0)
        const uint4* colp = reinterpret_cast<const uint4*>(
            Xbf + ((size_t)b * L_ + lt * 16 + mn) * KPAD_);
        BF16x16 bb0, bb1;
        bb0.q[0] = colp[hi];     bb0.q[1] = colp[2 + hi];    // K 0..31
        bb1.q[0] = colp[4 + hi]; bb1.q[1] = colp[6 + hi];    // K 32..63

        v8f c = {};
        c = __builtin_amdgcn_wmma_f32_16x16x32_bf16(false, a0.v, false, bb0.v, (short)0, c, false, false);
        c = __builtin_amdgcn_wmma_f32_16x16x32_bf16(false, a1.v, false, bb1.v, (short)0, c, false, false);

        size_t outbase = (size_t)b * OC_ * L_ + (size_t)lt * 16 + mn;
#pragma unroll
        for (int r = 0; r < 8; ++r) {
            int row = ot * 16 + r + 8 * hi;   // C/D layout: VGPR r → M = r + 8*hi
            bc_all[outbase + (size_t)row * L_] = f2bf(c[r] + bias[row]);
        }
    }
}

// ------------- 9/10. selective scan (dir 0: along W; dir 1: along H) -------------
// 16 lanes per (b,row,c) group hold the S states; sum over S via width-16 shuffles.
__global__ void k_scan(const float* __restrict__ u, const float* __restrict__ d,
                       const unsigned short* __restrict__ bc_all, const float* __restrict__ A,
                       const float* __restrict__ D, const float* __restrict__ dtb,
                       float* __restrict__ yo, int dir) {
    int tid = blockIdx.x * blockDim.x + threadIdx.x;
    if (tid >= B_ * 48 * C_ * S_) return;
    int s = tid & 15;
    int c = (tid >> 4) % C_;
    int r = ((tid >> 4) / C_) % 48;
    int b = tid / (S_ * C_ * 48);
    float Acs = A[c * S_ + s];
    float Dc = D[c], db = dtb[c];
    size_t ub = ((size_t)b * C_ + c) * L_;
    size_t Bb = ((((size_t)b * 2 + 0) * C_ + c) * S_ + s) * L_;
    size_t Cb = ((((size_t)b * 2 + 1) * C_ + c) * S_ + s) * L_;
    float state = 0.f;
    for (int t = 0; t < 48; ++t) {
        int l = dir ? (t * W_ + r) : (r * W_ + t);
        float dp = d[ub + l] + db;
        float dsp = (dp > 20.f) ? dp : log1pf(__expf(dp));   // softplus
        float uu = u[ub + l];
        float dA = __expf(dsp * Acs);
        state = dA * state + dsp * bf2f(bc_all[Bb + l]) * uu;
        float p = state * bf2f(bc_all[Cb + l]);
        p += __shfl_xor(p, 1, 16);
        p += __shfl_xor(p, 2, 16);
        p += __shfl_xor(p, 4, 16);
        p += __shfl_xor(p, 8, 16);
        if (s == 0) yo[ub + l] = p + Dc * uu;
    }
}

// ------------- 11. out stage-1: t2[b][m][l] = sum_c 0.5*(yh+yv)*out1_w -------------
__global__ void k_out1(const float* __restrict__ yh, const float* __restrict__ yv,
                       const float* __restrict__ w, float* __restrict__ t2) {
    int idx = blockIdx.x * blockDim.x + threadIdx.x;
    if (idx >= B_ * MID_ * L_) return;
    int l = idx % L_, m = (idx / L_) % MID_, b = idx / (MID_ * L_);
    size_t base = (size_t)b * C_ * L_ + l;
    float acc = 0.f;
    for (int c = 0; c < C_; ++c)
        acc += 0.5f * (yh[base + (size_t)c * L_] + yv[base + (size_t)c * L_]) * w[m * C_ + c];
    t2[idx] = acc;
}

// ------------- 12. out stage-2 + res_scale -------------
__global__ void k_out2(const float* __restrict__ t2, const float* __restrict__ w2,
                       const float* __restrict__ b2, const float* __restrict__ rs,
                       float* __restrict__ yout) {
    int idx = blockIdx.x * blockDim.x + threadIdx.x;
    if (idx >= B_ * C_ * L_) return;
    int l = idx % L_, co = (idx / L_) % C_, b = idx / (C_ * L_);
    const float* tp = t2 + (size_t)b * MID_ * L_ + l;
    float acc = b2[co];
    for (int m = 0; m < MID_; ++m)
        acc += tp[(size_t)m * L_] * w2[co * MID_ + m];
    yout[idx] = acc * rs[0];
}

// ------------- 14. final GN(y) + residual -------------
__global__ void k_final(const float* __restrict__ x, const float* __restrict__ yout,
                        const float* __restrict__ st, const float* __restrict__ gw,
                        const float* __restrict__ gb, float* __restrict__ out) {
    int idx = blockIdx.x * blockDim.x + threadIdx.x;
    if (idx >= B_ * C_ * L_) return;
    int c = (idx / L_) % C_, b = idx / (C_ * L_);
    float mu = st[b * 2 + 0], r = st[b * 2 + 1];
    float sc = r * gw[c], sh = gb[c] - mu * sc;
    out[idx] = x[idx] + yout[idx] * sc + sh;
}

extern "C" void kernel_launch(void* const* d_in, const int* in_sizes, int n_in,
                              void* d_out, int out_size, void* d_ws, size_t ws_size,
                              hipStream_t stream) {
    const float* x        = (const float*)d_in[0];
    const float* gn_pre_w = (const float*)d_in[1];
    const float* gn_pre_b = (const float*)d_in[2];
    const float* dw_w     = (const float*)d_in[3];
    const float* dw_b     = (const float*)d_in[4];
    const float* in1_w    = (const float*)d_in[5];
    const float* in2_w    = (const float*)d_in[6];
    const float* in2_b    = (const float*)d_in[7];
    const float* dt1_w    = (const float*)d_in[8];
    const float* dt2_w    = (const float*)d_in[9];
    const float* dt2_b    = (const float*)d_in[10];
    const float* bcb_w    = (const float*)d_in[11];
    const float* bch_w    = (const float*)d_in[12];
    const float* bch_b    = (const float*)d_in[13];
    const float* A        = (const float*)d_in[14];
    const float* D        = (const float*)d_in[15];
    const float* dt_bias  = (const float*)d_in[16];
    const float* out1_w   = (const float*)d_in[17];
    const float* out2_w   = (const float*)d_in[18];
    const float* out2_b   = (const float*)d_in[19];
    const float* res_sc   = (const float*)d_in[20];
    const float* gn_po_w  = (const float*)d_in[21];
    const float* gn_po_b  = (const float*)d_in[22];
    float* out = (float*)d_out;

    // ---- workspace carve-out (all chunk sizes multiples of 16 floats) ----
    const size_t N_BCL = (size_t)B_ * C_ * L_;   // 884736
    const size_t N_BML = (size_t)B_ * MID_ * L_; // 221184
    float* z    = (float*)d_ws;
    float* u    = z + N_BCL;
    float* dlt  = u + N_BCL;
    float* t1i  = dlt + N_BCL;
    float* t1d  = t1i + N_BML;
    float* bcm  = t1d + N_BML;
    float* t2   = bcm + N_BML;
    float* yh   = t2 + N_BML;
    float* yv   = yh + N_BCL;
    float* yout = yv + N_BCL;
    float* st   = yout + N_BCL;                  // [0..3] pre, [4..7] post
    unsigned short* Wbf   = (unsigned short*)(st + 16);
    unsigned short* Xbf   = Wbf + (size_t)OC_ * KPAD_;
    unsigned short* bcall = Xbf + (size_t)B_ * L_ * KPAD_;    // [B][OC_][L] bf16, 56.6 MB

    const int T = 256;
    const int g_bcl = (int)((N_BCL + T - 1) / T);  // 3456
    const int g_bml = (int)((N_BML + T - 1) / T);  // 864

    k_gn_stats<<<B_, T, 0, stream>>>(x, st);
    k_gn_conv_silu<<<g_bcl, T, 0, stream>>>(x, gn_pre_w, gn_pre_b, dw_w, dw_b, st, z);
    k_proj1<<<g_bml, T, 0, stream>>>(z, in1_w, dt1_w, t1i, t1d);
    k_bcmid<<<g_bml, T, 0, stream>>>(z, bcb_w, bcm);
    k_proj2<<<g_bcl, T, 0, stream>>>(t1i, t1d, in2_w, in2_b, dt2_w, dt2_b, u, dlt);
    k_cvt_w<<<(OC_ * KPAD_) / T, T, 0, stream>>>(bch_w, Wbf);
    k_cvt_x<<<(B_ * L_ * KPAD_) / T, T, 0, stream>>>(bcm, Xbf);
    {
        // grid: B * o-tiles * (l-tile chunks of 16)
        int blocks = B_ * (OC_ / 16) * ((L_ / 16) / 16);   // 2*384*9 = 6912
        k_bc_gemm_wmma<<<blocks, T, 0, stream>>>(Wbf, Xbf, bch_b, bcall);
    }
    {
        int threads = B_ * 48 * C_ * S_;          // 294912
        k_scan<<<threads / T, T, 0, stream>>>(u, dlt, bcall, A, D, dt_bias, yh, 0);
        k_scan<<<threads / T, T, 0, stream>>>(u, dlt, bcall, A, D, dt_bias, yv, 1);
    }
    k_out1<<<g_bml, T, 0, stream>>>(yh, yv, out1_w, t2);
    k_out2<<<g_bcl, T, 0, stream>>>(t2, out2_w, out2_b, res_sc, yout);
    k_gn_stats<<<B_, T, 0, stream>>>(yout, st + 4);
    k_final<<<g_bcl, T, 0, stream>>>(x, yout, st + 4, gn_po_w, gn_po_b, out);
}
